// RGCNBasisAttentionLayer_24627342475568
// MI455X (gfx1250) — compile-verified, matrix operand order checked by reading, not run on hardware
//
#include <hip/hip_runtime.h>
#include <hip/hip_bf16.h>

typedef __attribute__((ext_vector_type(16))) __bf16 v16bf;
typedef __attribute__((ext_vector_type(8)))  float  v8f;

#define NNODES 50000
#define NEDGES 400000
#define IN_F   32
#define OUT_F  32
#define NREL   32
#define NBASE  8
#define NTILES (NNODES / 16)   // 3125, exact

// ISA 7.12.2: 16-bit A-matrix 16x32 (and mirrored B 32x16) K index for
// (vgpr v, packed half p, lane-group grp = lane>>4).
__device__ __forceinline__ int kmap(int v, int p, int grp) {
    return ((v < 4) ? 0 : 16) + grp * 8 + ((v & 3) << 1) + p;
}

// ---------------------------------------------------------------------------
// K1: per-node dense GEMMs via WMMA.
//   HW[n, b*32+o] = sum_k h[n,k] * weight[b,k,o]      (8 bases x 2 col tiles)
//   out[n, o]     = sum_k h[n,k] * loop_weight[k,o]   (self-loop, 2 col tiles)
//   HA[n, b]      = sum_k h[n,k] * a_weight[b,k,0]    (1 padded tile)
// Also zeroes attnSum. One wave per 16-node row tile, 8 waves per block.
// ---------------------------------------------------------------------------
__global__ __launch_bounds__(256) void rgcn_node_gemm_kernel(
    const float* __restrict__ h, const float* __restrict__ weight,
    const float* __restrict__ a_weight, const float* __restrict__ loop_weight,
    float* __restrict__ HW, float* __restrict__ HA,
    float* __restrict__ attnSum, float* __restrict__ out)
{
    int gid = blockIdx.x * blockDim.x + threadIdx.x;
    if (gid < NNODES) attnSum[gid] = 0.0f;           // fresh every launch

    int tile = blockIdx.x * 8 + (threadIdx.x >> 5);  // wave-uniform
    if (tile >= NTILES) return;

    const int lane = threadIdx.x & 31;
    const int m    = lane & 15;   // row (A) / col (B,C,D) within tile
    const int grp  = lane >> 4;

    // ---- A tile: 16 rows of h, bf16 ----
    const int rowA = tile * 16 + m;
    v16bf A;
    #pragma unroll
    for (int v = 0; v < 8; ++v)
        #pragma unroll
        for (int p = 0; p < 2; ++p)
            A[2 * v + p] = (__bf16)h[rowA * IN_F + kmap(v, p, grp)];

    // ---- HW = h @ weight[b], for all 8 bases ----
    #pragma unroll
    for (int b = 0; b < NBASE; ++b) {
        #pragma unroll
        for (int c = 0; c < 2; ++c) {
            const int col = c * 16 + m;
            v16bf Bm;
            #pragma unroll
            for (int v = 0; v < 8; ++v)
                #pragma unroll
                for (int p = 0; p < 2; ++p)
                    Bm[2 * v + p] = (__bf16)weight[b * IN_F * OUT_F +
                                                   kmap(v, p, grp) * OUT_F + col];
            v8f acc = {};
            acc = __builtin_amdgcn_wmma_f32_16x16x32_bf16(
                false, A, false, Bm, (short)0, acc, false, false);
            #pragma unroll
            for (int v = 0; v < 8; ++v) {
                const int r = tile * 16 + v + 8 * grp;
                HW[r * (NBASE * OUT_F) + b * OUT_F + col] = acc[v];
            }
        }
    }

    // ---- self-loop: out = h @ loop_weight (accumulation base for agg) ----
    #pragma unroll
    for (int c = 0; c < 2; ++c) {
        const int col = c * 16 + m;
        v16bf Bm;
        #pragma unroll
        for (int v = 0; v < 8; ++v)
            #pragma unroll
            for (int p = 0; p < 2; ++p)
                Bm[2 * v + p] = (__bf16)loop_weight[kmap(v, p, grp) * OUT_F + col];
        v8f acc = {};
        acc = __builtin_amdgcn_wmma_f32_16x16x32_bf16(
            false, A, false, Bm, (short)0, acc, false, false);
        #pragma unroll
        for (int v = 0; v < 8; ++v) {
            const int r = tile * 16 + v + 8 * grp;
            out[r * OUT_F + col] = acc[v];
        }
    }

    // ---- HA = h @ a_weight (8 real columns, padded to a 16-wide tile) ----
    {
        const int col = m;  // column b of HA for col < 8
        v16bf Bm;
        #pragma unroll
        for (int v = 0; v < 8; ++v)
            #pragma unroll
            for (int p = 0; p < 2; ++p) {
                const int k = kmap(v, p, grp);
                Bm[2 * v + p] = (col < NBASE) ? (__bf16)a_weight[col * IN_F + k]
                                              : (__bf16)0.0f;
            }
        v8f acc = {};
        acc = __builtin_amdgcn_wmma_f32_16x16x32_bf16(
            false, A, false, Bm, (short)0, acc, false, false);
        if (col < NBASE) {
            #pragma unroll
            for (int v = 0; v < 8; ++v) {
                const int r = tile * 16 + v + 8 * grp;
                HA[r * NBASE + col] = acc[v];
            }
        }
    }
}

// ---------------------------------------------------------------------------
// K2: per-edge attention logit + exp(tanh) + segment-sum over source nodes.
// ---------------------------------------------------------------------------
__global__ __launch_bounds__(256) void rgcn_edge_atten_kernel(
    const float* __restrict__ HA, const float* __restrict__ w_comp_a,
    const int* __restrict__ esrc, const int* __restrict__ etype,
    float* __restrict__ atten, float* __restrict__ attnSum)
{
    int e = blockIdx.x * blockDim.x + threadIdx.x;
    if (e >= NEDGES) return;
    const int s = esrc[e];
    const int t = etype[e];
    float logit = 0.0f;
    #pragma unroll
    for (int b = 0; b < NBASE; ++b)
        logit += w_comp_a[t * NBASE + b] * HA[s * NBASE + b];
    const float a = __expf(tanhf(logit));
    atten[e] = a;
    atomicAdd(&attnSum[s], a);
}

// ---------------------------------------------------------------------------
// K3: one wave per edge; lane o produces output feature o.
//   msg[e,o] = (atten/attnSum[src]) * norm[e] * sum_b w_comp[t,b]*HW[src,b,o]
// scattered into out[dst,:] (which already holds the self-loop term).
// ---------------------------------------------------------------------------
__global__ __launch_bounds__(256) void rgcn_edge_msg_kernel(
    const float* __restrict__ HW, const float* __restrict__ w_comp,
    const float* __restrict__ norm, const float* __restrict__ atten,
    const float* __restrict__ attnSum,
    const int* __restrict__ esrc, const int* __restrict__ edst,
    const int* __restrict__ etype, float* __restrict__ out)
{
    int e = blockIdx.x * 8 + (threadIdx.x >> 5);
    if (e >= NEDGES) return;
    const int o   = threadIdx.x & 31;
    const int s   = esrc[e];
    const int d   = edst[e];
    const int t   = etype[e];
    const float coeff = (atten[e] / attnSum[s]) * norm[e];
    float acc = 0.0f;
    #pragma unroll
    for (int b = 0; b < NBASE; ++b)
        acc += w_comp[t * NBASE + b] * HW[s * (NBASE * OUT_F) + b * OUT_F + o];
    atomicAdd(&out[d * OUT_F + o], acc * coeff);
}

// ---------------------------------------------------------------------------
// K4: final ReLU over out = agg + h @ loop_weight.
// ---------------------------------------------------------------------------
__global__ __launch_bounds__(256) void rgcn_relu_kernel(float* __restrict__ out)
{
    int i = blockIdx.x * blockDim.x + threadIdx.x;
    if (i < NNODES * OUT_F) out[i] = fmaxf(out[i], 0.0f);
}

extern "C" void kernel_launch(void* const* d_in, const int* in_sizes, int n_in,
                              void* d_out, int out_size, void* d_ws, size_t ws_size,
                              hipStream_t stream) {
    (void)in_sizes; (void)n_in; (void)out_size; (void)ws_size;
    const float* h           = (const float*)d_in[0];
    const float* norm        = (const float*)d_in[1];
    const float* weight      = (const float*)d_in[2];
    const float* a_weight    = (const float*)d_in[3];
    const float* w_comp      = (const float*)d_in[4];
    const float* w_comp_a    = (const float*)d_in[5];
    const float* loop_weight = (const float*)d_in[6];
    const int*   edge_src    = (const int*)d_in[7];
    const int*   edge_dst    = (const int*)d_in[8];
    const int*   edge_type   = (const int*)d_in[9];
    float* out = (float*)d_out;

    // Workspace carve-up (floats): HW | HA | attnSum | atten
    float* HW      = (float*)d_ws;                          // N * 256
    float* HA      = HW + (size_t)NNODES * NBASE * OUT_F;   // N * 8
    float* attnSum = HA + (size_t)NNODES * NBASE;           // N
    float* atten   = attnSum + (size_t)NNODES;              // E

    // K1: dense node GEMMs via WMMA (HW, HA, self-loop into out; zero attnSum)
    rgcn_node_gemm_kernel<<<(NTILES + 7) / 8, 256, 0, stream>>>(
        h, weight, a_weight, loop_weight, HW, HA, attnSum, out);

    // K2: per-edge attention + segment sum over src
    rgcn_edge_atten_kernel<<<(NEDGES + 255) / 256, 256, 0, stream>>>(
        HA, w_comp_a, edge_src, edge_type, atten, attnSum);

    // K3: per-edge message + scatter to dst
    rgcn_edge_msg_kernel<<<(NEDGES + 7) / 8, 256, 0, stream>>>(
        HW, w_comp, norm, atten, attnSum, edge_src, edge_dst, edge_type, out);

    // K4: ReLU
    rgcn_relu_kernel<<<(NNODES * OUT_F + 255) / 256, 256, 0, stream>>>(out);
}
